// BertAttention_5128190951721
// MI455X (gfx1250) — matrix operand order repaired and checked by reading, not compiled, not run on hardware
//
#include <hip/hip_runtime.h>
#include <hip/hip_bf16.h>

// ---- problem dims (fixed by the reference) ----
#define SS 16384
#define DDIM 1024

typedef __attribute__((ext_vector_type(16))) __bf16       v16bf;
typedef __attribute__((ext_vector_type(8)))  float        v8f;
typedef __attribute__((ext_vector_type(4)))  unsigned int v4u;

union FragU { v16bf v; v4u u[2]; };

__device__ __forceinline__ unsigned short f2bf(float f) {
  unsigned u = __float_as_uint(f);
  u += 0x7FFFu + ((u >> 16) & 1u);            // round-to-nearest-even
  return (unsigned short)(u >> 16);
}

// ---------------- fp32 -> bf16 cast ----------------
__global__ void cast_bf16_kernel(const float* __restrict__ in,
                                 unsigned short* __restrict__ out, long n) {
  long i = (long)blockIdx.x * blockDim.x + threadIdx.x;
  if (i < n) out[i] = f2bf(in[i]);
}

// ---------------- bf16 tile transpose [R,C] -> [C,R] ----------------
__global__ __launch_bounds__(256)
void transpose_bf16_kernel(const unsigned short* __restrict__ in,
                           unsigned short* __restrict__ out, int R, int C) {
  __shared__ unsigned short tile[32][33];
  const int cT = blockIdx.x * 32, rT = blockIdx.y * 32;
  const int tx = threadIdx.x, ty = threadIdx.y;     // block (32,8)
#pragma unroll
  for (int i = 0; i < 4; ++i)
    tile[ty + i * 8][tx] = in[(long)(rT + ty + i * 8) * C + (cT + tx)];
  __syncthreads();
#pragma unroll
  for (int i = 0; i < 4; ++i)
    out[(long)(cT + ty + i * 8) * R + (rT + tx)] = tile[tx][ty + i * 8];
}

// ---------------- NT GEMM via v_wmma_f32_16x16x32_bf16 ----------------
// C[M,N] = scale * (A[M,K] @ B[N,K]^T) + bias
// block = 256 threads = 8 waves arranged 2(M) x 4(N), each wave owns 64x64
template<bool OUT_BF16>
__global__ __launch_bounds__(256)
void gemm_nt_wmma(const unsigned short* __restrict__ A,
                  const unsigned short* __restrict__ B,
                  const float* __restrict__ bias,
                  void* __restrict__ Cp,
                  int M, int N, int K, float scale) {
  const int lane  = threadIdx.x & 31;
  const int wave  = threadIdx.x >> 5;
  const int waveM = wave >> 2;                // 0..1
  const int waveN = wave & 3;                 // 0..3
  const long mBase = (long)blockIdx.y * 128 + waveM * 64;
  const long nBase = (long)blockIdx.x * 256 + waveN * 64;
  const int lh = lane >> 4;                   // half-wave select
  const int ll = lane & 15;

  const v8f zero = {0.f, 0.f, 0.f, 0.f, 0.f, 0.f, 0.f, 0.f};
  v8f c[4][4];
#pragma unroll
  for (int i = 0; i < 4; ++i)
#pragma unroll
    for (int j = 0; j < 4; ++j) c[i][j] = zero;

  // Per-lane fragment base pointers (K-contiguous chunks per the ISA
  // 16-bit A 16x32 / B 32x16 VGPR striping).
  const unsigned short* aBase[4];
  const unsigned short* bBase[4];
#pragma unroll
  for (int i = 0; i < 4; ++i)
    aBase[i] = A + (mBase + i * 16 + ll) * (long)K + lh * 8;   // K {0..7,16..23} | {8..15,24..31}
#pragma unroll
  for (int j = 0; j < 4; ++j)
    bBase[j] = B + (nBase + j * 16 + ll) * (long)K + lh * 8;   // K {0..7,16..23} | {8..15,24..31}

  for (int k0 = 0; k0 < K; k0 += 32) {
    FragU a[4], b[4];
#pragma unroll
    for (int i = 0; i < 4; ++i) {
      a[i].u[0] = *(const v4u*)(aBase[i] + k0);
      a[i].u[1] = *(const v4u*)(aBase[i] + k0 + 16);
    }
#pragma unroll
    for (int j = 0; j < 4; ++j) {
      b[j].u[0] = *(const v4u*)(bBase[j] + k0);
      b[j].u[1] = *(const v4u*)(bBase[j] + k0 + 16);
    }
    // speculative prefetch of next K-slab (lowers to global_prefetch)
    __builtin_prefetch(aBase[0] + k0 + 32, 0, 0);
    __builtin_prefetch(bBase[0] + k0 + 32, 0, 0);
#pragma unroll
    for (int i = 0; i < 4; ++i)
#pragma unroll
      for (int j = 0; j < 4; ++j)
        c[i][j] = __builtin_amdgcn_wmma_f32_16x16x32_bf16(
            false, a[i].v, false, b[j].v, (short)0, c[i][j], false, false);
  }

  // Epilogue: C/D layout — lane l, vgpr r -> (m = r + 8*(l>>4), n = l&15)
#pragma unroll
  for (int i = 0; i < 4; ++i) {
#pragma unroll
    for (int j = 0; j < 4; ++j) {
      const long mrow = mBase + i * 16 + 8 * lh;
      const int  ncol = (int)nBase + j * 16 + ll;
      const float badd = bias ? bias[ncol] : 0.f;
#pragma unroll
      for (int r = 0; r < 8; ++r) {
        const float val = c[i][j][r] * scale + badd;
        const long idx = (mrow + r) * (long)N + ncol;
        if (OUT_BF16) ((unsigned short*)Cp)[idx] = f2bf(val);
        else          ((float*)Cp)[idx] = val;
      }
    }
  }
}

// ---------------- row softmax: scores[1024x1024] f32 -> attn f32 + attn^T bf16 ----------------
__global__ __launch_bounds__(256)
void softmax_rows_kernel(const float* __restrict__ sc,
                         float* __restrict__ attn,
                         unsigned short* __restrict__ attnT, int Dn) {
  const int row = blockIdx.x;
  const int tid = threadIdx.x;                 // 256 threads, Dn = 1024
  __shared__ float red[256];
  const float* r = sc + (long)row * Dn;

  float mx = -3.402823466e+38f;
  for (int cidx = tid; cidx < Dn; cidx += 256) mx = fmaxf(mx, r[cidx]);
  red[tid] = mx; __syncthreads();
  for (int s = 128; s > 0; s >>= 1) {
    if (tid < s) red[tid] = fmaxf(red[tid], red[tid + s]);
    __syncthreads();
  }
  mx = red[0]; __syncthreads();

  float sum = 0.f;
  for (int cidx = tid; cidx < Dn; cidx += 256) sum += expf(r[cidx] - mx);
  red[tid] = sum; __syncthreads();
  for (int s = 128; s > 0; s >>= 1) {
    if (tid < s) red[tid] += red[tid + s];
    __syncthreads();
  }
  const float inv = 1.f / red[0];

  for (int cidx = tid; cidx < Dn; cidx += 256) {
    const float a = expf(r[cidx] - mx) * inv;
    attn[(long)row * Dn + cidx] = a;
    attnT[(long)cidx * Dn + row] = f2bf(a);
  }
}

// ---------------- driver ----------------
extern "C" void kernel_launch(void* const* d_in, const int* in_sizes, int n_in,
                              void* d_out, int out_size, void* d_ws, size_t ws_size,
                              hipStream_t stream) {
  const float* x  = (const float*)d_in[0];
  const float* Wq = (const float*)d_in[1];
  const float* bq = (const float*)d_in[2];
  const float* Wk = (const float*)d_in[3];
  const float* bk = (const float*)d_in[4];
  const float* Wv = (const float*)d_in[5];
  const float* bv = (const float*)d_in[6];
  float* out = (float*)d_out;                       // [S*D] out ++ [D*D] attn

  const long SD = (long)SS * DDIM;                  // 16,777,216
  const long DDn = (long)DDIM * DDIM;               // 1,048,576

  // workspace layout (bytes)
  char* ws = (char*)d_ws;
  unsigned short* xb  = (unsigned short*)(ws);                      // 32 MB (reused as qT later)
  unsigned short* wqb = (unsigned short*)(ws + SD * 2);             // 2 MB
  unsigned short* wkb = wqb + DDn;                                  // 2 MB
  unsigned short* wvb = wkb + DDn;                                  // 2 MB
  unsigned short* qb  = wvb + DDn;                                  // 32 MB
  unsigned short* kb  = qb + SD;                                    // 32 MB
  unsigned short* vb  = kb + SD;                                    // 32 MB
  unsigned short* kT  = vb + SD;                                    // 32 MB
  float*          scores = (float*)(kT + SD);                       // 4 MB
  unsigned short* attnT  = (unsigned short*)(scores + DDn);         // 2 MB
  unsigned short* qT = xb;   // x is dead after the three QKV GEMMs

  // 1) casts to bf16
  cast_bf16_kernel<<<(int)(SD / 256), 256, 0, stream>>>(x, xb, SD);
  cast_bf16_kernel<<<(int)(DDn / 256), 256, 0, stream>>>(Wq, wqb, DDn);
  cast_bf16_kernel<<<(int)(DDn / 256), 256, 0, stream>>>(Wk, wkb, DDn);
  cast_bf16_kernel<<<(int)(DDn / 256), 256, 0, stream>>>(Wv, wvb, DDn);

  // 2) q/k/v = x @ W^T + b   (NT GEMM, bf16 out)
  dim3 gQKV(DDIM / 256, SS / 128);
  gemm_nt_wmma<true><<<gQKV, 256, 0, stream>>>(xb, wqb, bq, qb, SS, DDIM, DDIM, 1.f);
  gemm_nt_wmma<true><<<gQKV, 256, 0, stream>>>(xb, wkb, bk, kb, SS, DDIM, DDIM, 1.f);
  gemm_nt_wmma<true><<<gQKV, 256, 0, stream>>>(xb, wvb, bv, vb, SS, DDIM, DDIM, 1.f);

  // 3) transposes for the feature-axis score GEMM
  dim3 tBlk(32, 8);
  dim3 tGrd(DDIM / 32, SS / 32);
  transpose_bf16_kernel<<<tGrd, tBlk, 0, stream>>>(qb, qT, SS, DDIM);
  transpose_bf16_kernel<<<tGrd, tBlk, 0, stream>>>(kb, kT, SS, DDIM);

  // 4) scores[d,e] = (1/sqrt(D)) * sum_s q[s,d] k[s,e]  == qT @ kT^T (NT)
  dim3 gSC(DDIM / 256, DDIM / 128);
  gemm_nt_wmma<false><<<gSC, 256, 0, stream>>>(qT, kT, nullptr, scores,
                                               DDIM, DDIM, SS, 0.03125f /* 1/sqrt(1024) */);

  // 5) softmax rows -> attn (f32, second output) + attn^T (bf16 for final GEMM)
  softmax_rows_kernel<<<DDIM, 256, 0, stream>>>(scores, out + SD, attnT, DDIM);

  // 6) out = v @ attn  == v @ (attnT)^T (NT, f32 out)
  dim3 gOUT(DDIM / 256, SS / 128);
  gemm_nt_wmma<false><<<gOUT, 256, 0, stream>>>(vb, attnT, nullptr, out,
                                                SS, DDIM, DDIM, 1.f);
}
